// CausalSelfAttention_51728586113453
// MI455X (gfx1250) — compile-verified, matrix operand order checked by reading, not run on hardware
//
#include <hip/hip_runtime.h>
#include <hip/hip_bf16.h>

// ---------------- problem constants (from reference) ----------------
#define BATCH   4
#define SEQ     2048
#define CDIM    2048
#define N_HEAD  16
#define N_KV    4
#define HD      128
#define REP     (N_HEAD / N_KV)   // 4

typedef __attribute__((ext_vector_type(16))) __bf16 bf16x16;
typedef __attribute__((ext_vector_type(8)))  float  f32x8;

union FragU { uint4 q[2]; bf16x16 v; };
union U4H  { uint4 q; __bf16 h[8]; };
union U2H  { uint2 u; __bf16 h[4]; };

static __device__ __forceinline__ bf16x16 load_frag(const __bf16* p0, const __bf16* p1) {
  FragU u;
  u.q[0] = *(const uint4*)p0;
  u.q[1] = *(const uint4*)p1;
  return u.v;
}

static __device__ __forceinline__ f32x8 wmma_bf16(bf16x16 a, bf16x16 b, f32x8 c) {
  // D = A(16x32) * B(32x16) + C, fp32 accumulate
  return __builtin_amdgcn_wmma_f32_16x16x32_bf16(false, a, false, b, (short)0, c, false, false);
}

static __device__ __forceinline__ f32x8 zero8() {
  f32x8 z;
  #pragma unroll
  for (int i = 0; i < 8; ++i) z[i] = 0.0f;
  return z;
}

// ---- gfx1250 async global->LDS copy (ASYNCcnt-tracked) via inline asm ----
// GV mode: global_load_async_to_lds_b128 vdst(LDS byte addr), vaddr(64b), off
// Generic pointers to __shared__ carry the LDS offset in their low 32 bits.
#if defined(__gfx1250__)
static __device__ __forceinline__ void async_copy16(const void* g, void* l) {
  unsigned           lds = (unsigned)(size_t)l;
  unsigned long long ga  = (unsigned long long)(size_t)g;
  asm volatile("global_load_async_to_lds_b128 %0, %1, off"
               :: "v"(lds), "v"(ga)
               : "memory");
}
static __device__ __forceinline__ void async_wait0() {
  asm volatile("s_wait_asynccnt 0x0" ::: "memory");
}
#else
static __device__ __forceinline__ void async_copy16(const void* g, void* l) {
  *(uint4*)l = *(const uint4*)g;
}
static __device__ __forceinline__ void async_wait0() {}
#endif

// ---------------- fp32 -> bf16 convert (vectorized) ----------------
__global__ void cvt_kernel(const float4* __restrict__ src, uint2* __restrict__ dst, long n4) {
  long i = (long)blockIdx.x * blockDim.x + threadIdx.x;
  if (i < n4) {
    float4 v = src[i];
    U2H u;
    u.h[0] = (__bf16)v.x; u.h[1] = (__bf16)v.y;
    u.h[2] = (__bf16)v.z; u.h[3] = (__bf16)v.w;
    dst[i] = u.u;
  }
}

// ---------------- RoPE: fp32 in -> (optional fp32 out) + bf16 out ----------------
__global__ void rope_kernel(const float* __restrict__ src, const float* __restrict__ cs,
                            const float* __restrict__ sn, float* __restrict__ outF,
                            __bf16* __restrict__ outB, int nh, long npairs) {
  long p = (long)blockIdx.x * blockDim.x + threadIdx.x;
  if (p >= npairs) return;
  int  i  = (int)(p & 63);          // pair index within head dim (half = 64)
  long r  = p >> 6;                 // ((b*T + t)*nh + h)
  long bt = r / nh;                 // b*T + t
  int  t  = (int)(bt % SEQ);
  size_t base = (size_t)r * HD + 2 * (size_t)i;
  float xr = src[base], xi = src[base + 1];
  float c = cs[(size_t)t * 64 + i], s = sn[(size_t)t * 64 + i];
  float orr = xr * c - xi * s;
  float oii = xr * s + xi * c;
  if (outF) { outF[base] = orr; outF[base + 1] = oii; }
  outB[base]     = (__bf16)orr;
  outB[base + 1] = (__bf16)oii;
}

// ---------------- V passthrough: fp32 copy + bf16 ----------------
__global__ void copyv_kernel(const float* __restrict__ src, float* __restrict__ outF,
                             __bf16* __restrict__ outB, long n) {
  long i = (long)blockIdx.x * blockDim.x + threadIdx.x;
  if (i < n) { float v = src[i]; outF[i] = v; outB[i] = (__bf16)v; }
}

// ---------------- WMMA GEMM: C(MxN fp32) = A(MxK bf16) * B(KxN bf16) ----------------
// Block tile 128x128, K-step 32 (one WMMA depth). 256 threads = 8 waves,
// each wave computes a 32x64 sub-tile (2x4 grid of 16x16 accumulators).
// Double-buffered LDS; A tiles staged with ASYNCcnt-tracked async copies.
__global__ __launch_bounds__(256) void gemm_bf16(const __bf16* __restrict__ A,
                                                 const __bf16* __restrict__ Bm,
                                                 float* __restrict__ C,
                                                 int M, int N, int K) {
  __shared__ __align__(16) __bf16 As[2][128 * 32];   // row-major, 2 x 8 KB
  __shared__ __align__(16) __bf16 Bt[2][128 * 32];   // B transposed (N x K), 2 x 8 KB
  const int tid    = threadIdx.x;
  const int lane   = tid & 31;
  const int wave   = tid >> 5;
  const int laneLo = lane & 15;
  const int laneHi = lane >> 4;        // 0 or 1
  const int tileM  = blockIdx.y * 128;
  const int tileN  = blockIdx.x * 128;
  const int wRow   = (wave >> 1) * 32;
  const int wCol   = (wave & 1) * 64;

  f32x8 acc[2][4];
  #pragma unroll
  for (int r = 0; r < 2; ++r)
    #pragma unroll
    for (int c = 0; c < 4; ++c) acc[r][c] = zero8();

  const int kb = laneHi * 8;           // per-lane K sub-offset (ISA 16-bit A layout)

  // ---- prologue: stage tile 0 ----
  {
    #pragma unroll
    for (int i = 0; i < 2; ++i) {
      int uid = tid + i * 256;         // 512 x 16B for the A tile
      int row = uid >> 2;
      int c8  = (uid & 3) << 3;
      async_copy16(&A[(size_t)(tileM + row) * K + c8], &As[0][row * 32 + c8]);
    }
    #pragma unroll
    for (int i = 0; i < 2; ++i) {
      int uid  = tid + i * 256;        // 512 x 16B for the B tile
      int krow = uid >> 4;
      int n8   = (uid & 15) << 3;
      U4H u;
      u.q = *(const uint4*)&Bm[(size_t)krow * N + tileN + n8];
      #pragma unroll
      for (int j = 0; j < 8; ++j) Bt[0][(n8 + j) * 32 + krow] = u.h[j];
    }
    async_wait0();
    __syncthreads();
  }

  for (int kt = 0; kt < K; kt += 32) {
    const int  buf  = (kt >> 5) & 1;
    const int  nxt  = buf ^ 1;
    const bool more = (kt + 32) < K;

    U4H breg[2];
    if (more) {
      // issue next A tile via async DMA into the other buffer
      #pragma unroll
      for (int i = 0; i < 2; ++i) {
        int uid = tid + i * 256;
        int row = uid >> 2;
        int c8  = (uid & 3) << 3;
        async_copy16(&A[(size_t)(tileM + row) * K + kt + 32 + c8],
                     &As[nxt][row * 32 + c8]);
      }
      // fetch next B tile into registers (needs transpose on LDS store)
      #pragma unroll
      for (int i = 0; i < 2; ++i) {
        int uid  = tid + i * 256;
        int krow = uid >> 4;
        int n8   = (uid & 15) << 3;
        breg[i].q = *(const uint4*)&Bm[(size_t)(kt + 32 + krow) * N + tileN + n8];
      }
      if (kt + 64 < K) {   // L2 prefetch for the tile after next
        __builtin_prefetch(&A[(size_t)(tileM + (tid >> 1)) * K + kt + 64], 0, 1);
        __builtin_prefetch(&Bm[(size_t)(kt + 64 + (tid & 31)) * N + tileN], 0, 1);
      }
    }

    // ---- compute on current buffer ----
    bf16x16 aF[2], bF[4];
    #pragma unroll
    for (int r = 0; r < 2; ++r) {
      int row = wRow + r * 16 + laneLo;
      aF[r] = load_frag(&As[buf][row * 32 + kb], &As[buf][row * 32 + kb + 16]);
    }
    #pragma unroll
    for (int c = 0; c < 4; ++c) {
      int col = wCol + c * 16 + laneLo;
      bF[c] = load_frag(&Bt[buf][col * 32 + kb], &Bt[buf][col * 32 + kb + 16]);
    }
    #pragma unroll
    for (int r = 0; r < 2; ++r)
      #pragma unroll
      for (int c = 0; c < 4; ++c)
        acc[r][c] = wmma_bf16(aF[r], bF[c], acc[r][c]);

    if (more) {
      #pragma unroll
      for (int i = 0; i < 2; ++i) {
        int uid  = tid + i * 256;
        int krow = uid >> 4;
        int n8   = (uid & 15) << 3;
        #pragma unroll
        for (int j = 0; j < 8; ++j) Bt[nxt][(n8 + j) * 32 + krow] = breg[i].h[j];
      }
    }
    async_wait0();
    __syncthreads();
  }

  // store: C/D layout — VGPR j, lanes 0-15 -> row j, lanes 16-31 -> row j+8
  #pragma unroll
  for (int r = 0; r < 2; ++r)
    #pragma unroll
    for (int c = 0; c < 4; ++c)
      #pragma unroll
      for (int j = 0; j < 8; ++j) {
        int row = tileM + wRow + r * 16 + j + laneHi * 8;
        int col = tileN + wCol + c * 16 + laneLo;
        C[(size_t)row * N + col] = acc[r][c][j];
      }
}

// ---------------- Flash attention (causal, GQA) ----------------
// Grid: (T/64, N_HEAD, BATCH). Block: 128 threads = 4 waves; wave w owns 16
// query rows. KV streamed in 64-key tiles through LDS; online softmax.
__global__ __launch_bounds__(128) void flash_attn(const __bf16* __restrict__ Q,
                                                  const __bf16* __restrict__ Kg,
                                                  const __bf16* __restrict__ Vg,
                                                  __bf16* __restrict__ Y) {
  __shared__ __align__(16) __bf16 Ks[64 * 128];       // keys x D        (16 KB)
  __shared__ __align__(16) __bf16 Vs[128 * 64];       // D x keys (V^T)  (16 KB)
  __shared__ __align__(16) __bf16 Ps[4][16 * 64];     // per-wave P scratch (8 KB)

  const int tid    = threadIdx.x;
  const int lane   = tid & 31;
  const int wave   = tid >> 5;
  const int laneLo = lane & 15;
  const int laneHi = lane >> 4;
  const int qtile  = blockIdx.x;
  const int h      = blockIdx.y;
  const int b      = blockIdx.z;
  const int g      = h / REP;
  const int qBase  = qtile * 64 + wave * 16;
  const float scale = 0.08838834764831845f;   // 1/sqrt(128)
  const float NEG   = -__builtin_inff();

  // Q fragments for this wave's 16 rows, full D=128 (4 chunks of K=32)
  bf16x16 qF[4];
  {
    const size_t qrow = ((size_t)(b * SEQ + qBase + laneLo) * N_HEAD + h) * HD;
    #pragma unroll
    for (int kc = 0; kc < 4; ++kc) {
      int off = kc * 32 + laneHi * 8;
      qF[kc] = load_frag(&Q[qrow + off], &Q[qrow + off + 16]);
    }
  }

  f32x8 Oacc[8];
  #pragma unroll
  for (int dt = 0; dt < 8; ++dt) Oacc[dt] = zero8();
  float m[8], l[8];
  #pragma unroll
  for (int j = 0; j < 8; ++j) { m[j] = NEG; l[j] = 0.0f; }

  for (int kt = 0; kt <= qtile; ++kt) {
    // stage K tile (async DMA) and transposed V tile (1024 x 16B over 128 thr)
    #pragma unroll
    for (int i = 0; i < 8; ++i) {
      int uid = tid + i * 128;
      int row = uid >> 4;              // key within tile
      int d8  = (uid & 15) << 3;
      size_t ga = ((size_t)(b * SEQ + kt * 64 + row) * N_KV + g) * HD + d8;
      async_copy16(&Kg[ga], &Ks[row * 128 + d8]);
      U4H u;
      u.q = *(const uint4*)&Vg[ga];
      #pragma unroll
      for (int j = 0; j < 8; ++j) Vs[(d8 + j) * 64 + row] = u.h[j];
    }
    async_wait0();
    __syncthreads();

    // S = Q * K^T   (16 x 64 per wave)
    f32x8 S[4];
    #pragma unroll
    for (int c = 0; c < 4; ++c) S[c] = zero8();
    #pragma unroll
    for (int c = 0; c < 4; ++c) {
      int key = c * 16 + laneLo;
      #pragma unroll
      for (int kc = 0; kc < 4; ++kc) {
        int off = kc * 32 + laneHi * 8;
        bf16x16 kF = load_frag(&Ks[key * 128 + off], &Ks[key * 128 + off + 16]);
        S[c] = wmma_bf16(qF[kc], kF, S[c]);
      }
    }

    // scale + causal mask (only diagonal tile needs masking)
    const bool diag = (kt == qtile);
    #pragma unroll
    for (int c = 0; c < 4; ++c)
      #pragma unroll
      for (int j = 0; j < 8; ++j) {
        float s = S[c][j] * scale;
        if (diag) {
          int qi = qBase + j + laneHi * 8;
          int kk = kt * 64 + c * 16 + laneLo;
          if (kk > qi) s = NEG;
        }
        S[c][j] = s;
      }

    // online softmax: rows live in VGPR j (+8 for lane>=16), N across 16 lanes
    #pragma unroll
    for (int j = 0; j < 8; ++j) {
      float rmax = fmaxf(fmaxf(S[0][j], S[1][j]), fmaxf(S[2][j], S[3][j]));
      rmax = fmaxf(rmax, __shfl_xor(rmax, 1, 32));
      rmax = fmaxf(rmax, __shfl_xor(rmax, 2, 32));
      rmax = fmaxf(rmax, __shfl_xor(rmax, 4, 32));
      rmax = fmaxf(rmax, __shfl_xor(rmax, 8, 32));
      float mn    = fmaxf(m[j], rmax);
      float alpha = __expf(m[j] - mn);
      float sum = 0.0f;
      #pragma unroll
      for (int c = 0; c < 4; ++c) {
        float p = __expf(S[c][j] - mn);
        S[c][j] = p;
        sum += p;
      }
      sum += __shfl_xor(sum, 1, 32);
      sum += __shfl_xor(sum, 2, 32);
      sum += __shfl_xor(sum, 4, 32);
      sum += __shfl_xor(sum, 8, 32);
      l[j] = l[j] * alpha + sum;
      m[j] = mn;
      #pragma unroll
      for (int dt = 0; dt < 8; ++dt) Oacc[dt][j] *= alpha;
    }

    // transpose P through LDS into A-fragment layout
    #pragma unroll
    for (int c = 0; c < 4; ++c)
      #pragma unroll
      for (int j = 0; j < 8; ++j) {
        int row = j + laneHi * 8;
        Ps[wave][row * 64 + c * 16 + laneLo] = (__bf16)S[c][j];
      }
    __syncthreads();

    // O += P * V   (P: 16x64 bf16, V: 64x128 via Vs = V^T)
    #pragma unroll
    for (int kc = 0; kc < 2; ++kc) {
      int off = kc * 32 + laneHi * 8;
      bf16x16 pF = load_frag(&Ps[wave][laneLo * 64 + off],
                             &Ps[wave][laneLo * 64 + off + 16]);
      #pragma unroll
      for (int dt = 0; dt < 8; ++dt) {
        int d = dt * 16 + laneLo;
        bf16x16 vF = load_frag(&Vs[d * 64 + off], &Vs[d * 64 + off + 16]);
        Oacc[dt] = wmma_bf16(pF, vF, Oacc[dt]);
      }
    }
    __syncthreads();
  }

  // epilogue: O / l -> bf16 into attention output (input of Wc GEMM)
  #pragma unroll
  for (int dt = 0; dt < 8; ++dt)
    #pragma unroll
    for (int j = 0; j < 8; ++j) {
      int qi  = qBase + j + laneHi * 8;
      int col = h * HD + dt * 16 + laneLo;
      float v = Oacc[dt][j] / l[j];
      Y[(size_t)(b * SEQ + qi) * CDIM + col] = (__bf16)v;
    }
}

// ---------------- host-side orchestration ----------------
extern "C" void kernel_launch(void* const* d_in, const int* in_sizes, int n_in,
                              void* d_out, int out_size, void* d_ws, size_t ws_size,
                              hipStream_t stream) {
  (void)in_sizes; (void)n_in; (void)out_size; (void)ws_size;
  const float* x  = (const float*)d_in[0];
  const float* fc = (const float*)d_in[1];
  const float* fs = (const float*)d_in[2];
  const float* Wq = (const float*)d_in[3];
  const float* Wk = (const float*)d_in[4];
  const float* Wv = (const float*)d_in[5];
  const float* Wc = (const float*)d_in[6];

  const long BT  = (long)BATCH * SEQ;   // 8192
  const long DKV = (long)N_KV * HD;     // 512

  char* w = (char*)d_ws;
  size_t off = 0;
  auto carve = [&](size_t bytes) {
    void* p = w + off;
    off = (off + bytes + 255) & ~(size_t)255;
    return p;
  };
  __bf16* xb  = (__bf16*)carve((size_t)BT * CDIM * 2);
  __bf16* wqb = (__bf16*)carve((size_t)CDIM * CDIM * 2);
  __bf16* wkb = (__bf16*)carve((size_t)CDIM * DKV * 2);
  __bf16* wvb = (__bf16*)carve((size_t)CDIM * DKV * 2);
  __bf16* wcb = (__bf16*)carve((size_t)CDIM * CDIM * 2);
  float*  qf  = (float*) carve((size_t)BT * CDIM * 4);
  float*  kf  = (float*) carve((size_t)BT * DKV * 4);
  float*  vf  = (float*) carve((size_t)BT * DKV * 4);
  __bf16* qb  = (__bf16*)carve((size_t)BT * CDIM * 2);
  __bf16* kb  = (__bf16*)carve((size_t)BT * DKV * 2);
  __bf16* vb  = (__bf16*)carve((size_t)BT * DKV * 2);
  __bf16* yb  = (__bf16*)carve((size_t)BT * CDIM * 2);

  float* outY = (float*)d_out;
  float* outK = outY + (size_t)BT * CDIM;
  float* outV = outK + (size_t)BT * DKV;

  auto cvt = [&](const float* s, __bf16* d, long n) {
    long n4 = n / 4;
    cvt_kernel<<<dim3((unsigned)((n4 + 255) / 256)), dim3(256), 0, stream>>>(
        (const float4*)s, (uint2*)d, n4);
  };
  cvt(x,  xb,  BT * (long)CDIM);
  cvt(Wq, wqb, (long)CDIM * CDIM);
  cvt(Wk, wkb, (long)CDIM * DKV);
  cvt(Wv, wvb, (long)CDIM * DKV);
  cvt(Wc, wcb, (long)CDIM * CDIM);

  gemm_bf16<<<dim3(CDIM / 128, BT / 128), 256, 0, stream>>>(xb, wqb, qf, (int)BT, CDIM, CDIM);
  gemm_bf16<<<dim3((int)(DKV / 128), BT / 128), 256, 0, stream>>>(xb, wkb, kf, (int)BT, (int)DKV, CDIM);
  gemm_bf16<<<dim3((int)(DKV / 128), BT / 128), 256, 0, stream>>>(xb, wvb, vf, (int)BT, (int)DKV, CDIM);

  long qp = BT * (long)N_HEAD * 64;
  rope_kernel<<<dim3((unsigned)((qp + 255) / 256)), 256, 0, stream>>>(qf, fc, fs, nullptr, qb, N_HEAD, qp);
  long kp = BT * (long)N_KV * 64;
  rope_kernel<<<dim3((unsigned)((kp + 255) / 256)), 256, 0, stream>>>(kf, fc, fs, outK, kb, N_KV, kp);
  long vn = BT * DKV;
  copyv_kernel<<<dim3((unsigned)((vn + 255) / 256)), 256, 0, stream>>>(vf, outV, vb, vn);

  flash_attn<<<dim3(SEQ / 64, N_HEAD, BATCH), 128, 0, stream>>>(qb, kb, vb, yb);

  gemm_bf16<<<dim3(CDIM / 128, BT / 128), 256, 0, stream>>>(yb, wcb, outY, (int)BT, CDIM, CDIM);
}
